// HardQuadTripletSOSRLoss_40767829573846
// MI455X (gfx1250) — compile-verified, matrix operand order checked by reading, not run on hardware
//
#include <hip/hip_runtime.h>
#include <hip/hip_bf16.h>

typedef __attribute__((ext_vector_type(16))) _Float16 v16h;
typedef __attribute__((ext_vector_type(8)))  float    v8f;

constexpr int BB = 4;
constexpr int NN = 1024;
constexpr int CC = 128;
constexpr int HH = 256;
constexpr int WW = 256;
constexpr int MM = 1024;          // 32x32 grid cells
constexpr float GRIDF  = 8.0f;
constexpr float MARGIN = 1.0f;
constexpr float BIGV   = 10.0f;
constexpr float EPSV   = 1e-8f;
constexpr float EXCL   = 1e30f;

// ---------------------------------------------------------------------------
// init accumulators
// ---------------------------------------------------------------------------
__global__ void k_init(float* acc) {
    int i = threadIdx.x;
    if (i < 16) acc[i] = 0.0f;
}

// ---------------------------------------------------------------------------
// K1: L2-normalize anchors kp1_desc -> a32 (f32) and a16 (f16)
// ---------------------------------------------------------------------------
__global__ void k_anchor(const float* __restrict__ kp1_desc,
                         float* __restrict__ a32, _Float16* __restrict__ a16) {
    int idx = blockIdx.x * blockDim.x + threadIdx.x;   // b*NN + n
    if (idx >= BB * NN) return;
    const float* src = kp1_desc + (size_t)idx * CC;
    float s = 0.0f;
    for (int c = 0; c < CC; ++c) { float v = src[c]; s += v * v; }
    float inv = 1.0f / (sqrtf(s) + EPSV);
    for (int c = 0; c < CC; ++c) {
        float v = src[c] * inv;
        a32[(size_t)idx * CC + c] = v;
        a16[(size_t)idx * CC + c] = (_Float16)v;
    }
}

// ---------------------------------------------------------------------------
// helper: bilinear sample one channel of desc2 (C,H,W layout)
// ---------------------------------------------------------------------------
__device__ inline float bilin(const float* dc, int y0, int x0, int y1, int x1,
                              float w00, float w01, float w10, float w11) {
    return dc[y0 * WW + x0] * w00 + dc[y0 * WW + x1] * w01 +
           dc[y1 * WW + x0] * w10 + dc[y1 * WW + x1] * w11;
}

// ---------------------------------------------------------------------------
// K2: wa = l2n(bilinear(desc2, w_kp1)); pos = ||a - wa||; vis; cnt
// ---------------------------------------------------------------------------
__global__ void k_wa(const float* __restrict__ kp1, const float* __restrict__ wkp,
                     const float* __restrict__ desc2, const float* __restrict__ homo,
                     const float* __restrict__ a32,
                     float* __restrict__ wa32, _Float16* __restrict__ wa16,
                     float* __restrict__ pos, float* __restrict__ vis,
                     float* __restrict__ acc) {
    int idx = blockIdx.x * blockDim.x + threadIdx.x;   // b*NN + n
    if (idx >= BB * NN) return;
    int b = idx / NN;

    float x = wkp[(size_t)idx * 2 + 0];
    float y = wkp[(size_t)idx * 2 + 1];
    x = fminf(fmaxf(x, 0.0f), (float)(WW - 1));
    y = fminf(fmaxf(y, 0.0f), (float)(HH - 1));
    float x0f = floorf(x), y0f = floorf(y);
    float fx = x - x0f, fy = y - y0f;
    int x0 = (int)x0f, y0 = (int)y0f;
    int x1 = min(x0 + 1, WW - 1), y1 = min(y0 + 1, HH - 1);
    float w00 = (1.0f - fy) * (1.0f - fx), w01 = (1.0f - fy) * fx;
    float w10 = fy * (1.0f - fx),          w11 = fy * fx;

    const float* dbase = desc2 + (size_t)b * CC * HH * WW;
    float s = 0.0f;
    for (int c = 0; c < CC; ++c) {
        float v = bilin(dbase + (size_t)c * HH * WW, y0, x0, y1, x1, w00, w01, w10, w11);
        s += v * v;
    }
    float inv = 1.0f / (sqrtf(s) + EPSV);

    const float* arow = a32 + (size_t)idx * CC;
    float psum = 0.0f;
    for (int c = 0; c < CC; ++c) {
        float v = bilin(dbase + (size_t)c * HH * WW, y0, x0, y1, x1, w00, w01, w10, w11) * inv;
        wa32[(size_t)idx * CC + c] = v;
        wa16[(size_t)idx * CC + c] = (_Float16)v;
        float d = arow[c] - v;
        psum += d * d;
    }
    pos[idx] = sqrtf(psum);

    // visibility via homography warp of kp1
    float kx = kp1[(size_t)idx * 2 + 0], ky = kp1[(size_t)idx * 2 + 1];
    const float* Hm = homo + (size_t)b * 9;
    float wx = Hm[0] * kx + Hm[1] * ky + Hm[2];
    float wy = Hm[3] * kx + Hm[4] * ky + Hm[5];
    float wz = Hm[6] * kx + Hm[7] * ky + Hm[8];
    float iz = 1.0f / (wz + EPSV);
    float px = wx * iz, py = wy * iz;
    float v = (px >= 0.0f && px <= (float)(WW - 1) &&
               py >= 0.0f && py <= (float)(HH - 1)) ? 1.0f : 0.0f;
    vis[idx] = v;
    atomicAdd(&acc[0], v);
}

// ---------------------------------------------------------------------------
// K3: gd = l2n(bilinear(desc2, grid centers)) -> f16
// ---------------------------------------------------------------------------
__global__ void k_gd(const float* __restrict__ desc2, _Float16* __restrict__ gd16) {
    int idx = blockIdx.x * blockDim.x + threadIdx.x;   // b*MM + m
    if (idx >= BB * MM) return;
    int b = idx / MM;
    int m = idx % MM;
    float x = ((m & 31) + 0.5f) * GRIDF;   // cx
    float y = ((m >> 5) + 0.5f) * GRIDF;   // cy
    float x0f = floorf(x), y0f = floorf(y);
    float fx = x - x0f, fy = y - y0f;
    int x0 = (int)x0f, y0 = (int)y0f;
    int x1 = min(x0 + 1, WW - 1), y1 = min(y0 + 1, HH - 1);
    float w00 = (1.0f - fy) * (1.0f - fx), w01 = (1.0f - fy) * fx;
    float w10 = fy * (1.0f - fx),          w11 = fy * fx;

    const float* dbase = desc2 + (size_t)b * CC * HH * WW;
    float s = 0.0f;
    for (int c = 0; c < CC; ++c) {
        float v = bilin(dbase + (size_t)c * HH * WW, y0, x0, y1, x1, w00, w01, w10, w11);
        s += v * v;
    }
    float inv = 1.0f / (sqrtf(s) + EPSV);
    for (int c = 0; c < CC; ++c) {
        float v = bilin(dbase + (size_t)c * HH * WW, y0, x0, y1, x1, w00, w01, w10, w11) * inv;
        gd16[(size_t)idx * CC + c] = (_Float16)v;
    }
}

// ---------------------------------------------------------------------------
// fragment loader: row-major f16 rows -> WMMA 16x16x32 A/B operand layout
//   lane & 15  -> row (A) / column (B)
//   lane >> 4  -> K-half (0: K 0..7(+16..23), 1: K 8..15(+24..31))
// each fragment = two contiguous 16B chunks -> b128 / ds_b128 loads
// ---------------------------------------------------------------------------
__device__ inline v16h load_frag(const _Float16* rowptr, int k0, int khalf) {
    v16h f;
#pragma unroll
    for (int i = 0; i < 8; ++i) {
        int k = k0 + ((i >> 2) << 4) + (khalf << 3) + ((i & 3) << 1);
        f[2 * i]     = rowptr[k];
        f[2 * i + 1] = rowptr[k + 1];
    }
    return f;
}

__device__ inline float dot2dist(float dot) {
    return sqrtf(fmaxf(2.0f - 2.0f * dot, 0.0f) + EPSV);
}

// ---------------------------------------------------------------------------
// async stage one contiguous 4KB B-tile (16 rows x 256B) into LDS:
// 8 x global_load_async_to_lds_b128, 512B per instruction across 32 lanes
// ---------------------------------------------------------------------------
__device__ inline void stage_tile_async(const _Float16* tilebase, uint32_t ldsbase,
                                        int lane) {
    uint64_t gbase = (uint64_t)(uintptr_t)tilebase + (uint32_t)(lane * 16);
    uint32_t lbase = ldsbase + (uint32_t)(lane * 16);
#pragma unroll
    for (int i = 0; i < 8; ++i) {
        uint32_t la = lbase + (uint32_t)(i * 512);
        uint64_t ga = gbase + (uint32_t)(i * 512);
        asm volatile("global_load_async_to_lds_b128 %0, %1, off"
                     :: "v"(la), "v"(ga) : "memory");
    }
}

// ---------------------------------------------------------------------------
// FOS: per (b, 16-row strip): neg = dist(a, gd) via WMMA, spatial mask,
//      per-row top-8 smallest, hinge sum -> acc[1]
// block = 128 threads (4 waves); B tiles double-buffered in LDS via async DMA
// ---------------------------------------------------------------------------
__launch_bounds__(128)
__global__ void k_fos(const _Float16* __restrict__ a16, const _Float16* __restrict__ gd16,
                      const float* __restrict__ pos, const float* __restrict__ vis,
                      const float* __restrict__ wkp, float* __restrict__ acc) {
    __shared__ float dist[16 * MM];                 // 64 KB strip of distances
    __shared__ _Float16 bstage[4][2][16 * CC];      // 32 KB: per-wave ping-pong B tiles

    int b     = blockIdx.y;
    int strip = blockIdx.x * 16;
    int lane  = threadIdx.x & 31;
    int wave  = threadIdx.x >> 5;
    int half  = lane >> 4;
    int lrow  = lane & 15;

    const _Float16* Arow = a16 + ((size_t)b * NN + strip + lrow) * CC;
    v16h afrag[4];
#pragma unroll
    for (int kk = 0; kk < 4; ++kk) afrag[kk] = load_frag(Arow, kk * 32, half);

    uint32_t ldsb[2];
    ldsb[0] = (uint32_t)(uintptr_t)&bstage[wave][0][0];
    ldsb[1] = (uint32_t)(uintptr_t)&bstage[wave][1][0];

    // wave w owns tiles w, w+4, ..., w+60 (16 tiles); prefetch first tile
    stage_tile_async(gd16 + ((size_t)b * MM + wave * 16) * CC, ldsb[0], lane);

    for (int it = 0; it < 16; ++it) {
        int tile = wave + it * 4;
        int cur  = it & 1;
        if (it + 1 < 16) {
            int ntile = wave + (it + 1) * 4;
            stage_tile_async(gd16 + ((size_t)b * MM + ntile * 16) * CC,
                             ldsb[cur ^ 1], lane);
            asm volatile("s_wait_asynccnt 0x8" ::: "memory");  // current tile done
        } else {
            asm volatile("s_wait_asynccnt 0x0" ::: "memory");
        }

        const _Float16* Brow = &bstage[wave][cur][lrow * CC];
        v8f c = {};
#pragma unroll
        for (int kk = 0; kk < 4; ++kk) {
            v16h bfrag = load_frag(Brow, kk * 32, half);
            c = __builtin_amdgcn_wmma_f32_16x16x32_f16(
                    false, afrag[kk], false, bfrag, (short)0, c, false, false);
        }
        // D layout: col = col0 + (lane&15); row = r + 8*(lane>=16)
        int col0 = tile * 16;
        int ncol = col0 + lrow;
        float gx = ((ncol & 31) + 0.5f) * GRIDF;
        float gy = ((ncol >> 5) + 0.5f) * GRIDF;
#pragma unroll
        for (int r = 0; r < 8; ++r) {
            int mrow = r + 8 * half;
            int grow = strip + mrow;
            float wx = wkp[((size_t)b * NN + grow) * 2 + 0];
            float wy = wkp[((size_t)b * NN + grow) * 2 + 1];
            float dx = wx - gx, dy = wy - gy;
            float sp2 = dx * dx + dy * dy;
            float d = dot2dist(c[r]);
            if (sp2 < (2.0f * GRIDF) * (2.0f * GRIDF)) d = BIGV;
            dist[mrow * MM + ncol] = d;
        }
    }
    __syncthreads();

    // top-8 smallest per row; wave w owns rows 4w..4w+3
    for (int rr = 0; rr < 4; ++rr) {
        int mrow = wave * 4 + rr;
        int grow = strip + mrow;
        float* rowd = dist + mrow * MM;
        float p = pos[(size_t)b * NN + grow];
        float fsum = 0.0f;
        for (int t = 0; t < 8; ++t) {
            float mv = EXCL; int mi = 0;
            for (int j = lane; j < MM; j += 32) {
                float v = rowd[j];
                if (v < mv) { mv = v; mi = j; }
            }
#pragma unroll
            for (int off = 16; off > 0; off >>= 1) {
                float ov = __shfl_xor(mv, off, 32);
                int   oi = __shfl_xor(mi, off, 32);
                if (ov < mv || (ov == mv && oi < mi)) { mv = ov; mi = oi; }
            }
            if (lane == 0) rowd[mi] = EXCL;     // exclude for next pass
            fsum += fmaxf(p - mv + MARGIN, 0.0f);
        }
        if (lane == 0)
            atomicAdd(&acc[1], fsum * vis[(size_t)b * NN + grow]);
    }
}

// ---------------------------------------------------------------------------
// SOS: d2 = dist(wa, wa) via WMMA with spatial/diag mask; per-row top-8
//      smallest with indices; d1 recomputed sparsely from a32 -> acc[2]
// ---------------------------------------------------------------------------
__launch_bounds__(128)
__global__ void k_sos(const _Float16* __restrict__ wa16, const float* __restrict__ a32,
                      const float* __restrict__ vis, const float* __restrict__ wkp,
                      float* __restrict__ acc) {
    __shared__ float dist[16 * NN];
    __shared__ _Float16 bstage[4][2][16 * CC];

    int b     = blockIdx.y;
    int strip = blockIdx.x * 16;
    int lane  = threadIdx.x & 31;
    int wave  = threadIdx.x >> 5;
    int half  = lane >> 4;
    int lrow  = lane & 15;

    const _Float16* Arow = wa16 + ((size_t)b * NN + strip + lrow) * CC;
    v16h afrag[4];
#pragma unroll
    for (int kk = 0; kk < 4; ++kk) afrag[kk] = load_frag(Arow, kk * 32, half);

    uint32_t ldsb[2];
    ldsb[0] = (uint32_t)(uintptr_t)&bstage[wave][0][0];
    ldsb[1] = (uint32_t)(uintptr_t)&bstage[wave][1][0];

    stage_tile_async(wa16 + ((size_t)b * NN + wave * 16) * CC, ldsb[0], lane);

    for (int it = 0; it < 16; ++it) {
        int tile = wave + it * 4;
        int cur  = it & 1;
        if (it + 1 < 16) {
            int ntile = wave + (it + 1) * 4;
            stage_tile_async(wa16 + ((size_t)b * NN + ntile * 16) * CC,
                             ldsb[cur ^ 1], lane);
            asm volatile("s_wait_asynccnt 0x8" ::: "memory");
        } else {
            asm volatile("s_wait_asynccnt 0x0" ::: "memory");
        }

        const _Float16* Brow = &bstage[wave][cur][lrow * CC];
        v8f c = {};
#pragma unroll
        for (int kk = 0; kk < 4; ++kk) {
            v16h bfrag = load_frag(Brow, kk * 32, half);
            c = __builtin_amdgcn_wmma_f32_16x16x32_f16(
                    false, afrag[kk], false, bfrag, (short)0, c, false, false);
        }
        int col0 = tile * 16;
        int jcol = col0 + lrow;
        float jx = wkp[((size_t)b * NN + jcol) * 2 + 0];
        float jy = wkp[((size_t)b * NN + jcol) * 2 + 1];
#pragma unroll
        for (int r = 0; r < 8; ++r) {
            int mrow = r + 8 * half;
            int grow = strip + mrow;
            float wx = wkp[((size_t)b * NN + grow) * 2 + 0];
            float wy = wkp[((size_t)b * NN + grow) * 2 + 1];
            float dx = wx - jx, dy = wy - jy;
            float sp2 = dx * dx + dy * dy;
            float d = dot2dist(c[r]);
            if (sp2 < (2.0f * GRIDF) * (2.0f * GRIDF) || jcol == grow) d = BIGV;
            dist[mrow * NN + jcol] = d;
        }
    }
    __syncthreads();

    for (int rr = 0; rr < 4; ++rr) {
        int mrow = wave * 4 + rr;
        int grow = strip + mrow;
        float* rowd = dist + mrow * NN;
        float vals[8]; int idxs[8];
        for (int t = 0; t < 8; ++t) {
            float mv = EXCL; int mi = 0;
            for (int j = lane; j < NN; j += 32) {
                float v = rowd[j];
                if (v < mv) { mv = v; mi = j; }
            }
#pragma unroll
            for (int off = 16; off > 0; off >>= 1) {
                float ov = __shfl_xor(mv, off, 32);
                int   oi = __shfl_xor(mi, off, 32);
                if (ov < mv || (ov == mv && oi < mi)) { mv = ov; mi = oi; }
            }
            if (lane == 0) rowd[mi] = EXCL;
            vals[t] = mv; idxs[t] = mi;      // broadcast copies in every lane
        }
        // sparse d1: fp32 dots at the 8 selected indices
        const float* arow = a32 + ((size_t)b * NN + grow) * CC;
        float ssum = 0.0f;
        for (int t = 0; t < 8; ++t) {
            const float* ajrow = a32 + ((size_t)b * NN + idxs[t]) * CC;
            float dp = 0.0f;
            for (int cc = lane; cc < CC; cc += 32) dp += arow[cc] * ajrow[cc];
#pragma unroll
            for (int off = 16; off > 0; off >>= 1) dp += __shfl_xor(dp, off, 32);
            float d1 = dot2dist(dp);
            float okf = (vals[t] < 0.5f * BIGV) ? 1.0f : 0.0f;
            float term = (d1 - vals[t]) * okf;
            ssum += term * term;
        }
        float sos_i = sqrtf(ssum + EPSV);
        if (lane == 0)
            atomicAdd(&acc[2], sos_i * vis[(size_t)b * NN + grow]);
    }
}

// ---------------------------------------------------------------------------
// finalize: out = fos/(cnt*8) + sos/cnt
// ---------------------------------------------------------------------------
__global__ void k_final(const float* __restrict__ acc, float* __restrict__ out) {
    if (blockIdx.x == 0 && threadIdx.x == 0) {
        float cnt = fmaxf(acc[0], 1.0f);
        out[0] = acc[1] / (cnt * 8.0f) + acc[2] / cnt;
    }
}

// ---------------------------------------------------------------------------
extern "C" void kernel_launch(void* const* d_in, const int* in_sizes, int n_in,
                              void* d_out, int out_size, void* d_ws, size_t ws_size,
                              hipStream_t stream) {
    const float* kp1      = (const float*)d_in[0];   // (B,N,2)
    const float* w_kp1    = (const float*)d_in[1];   // (B,N,2)
    const float* kp1_desc = (const float*)d_in[2];   // (B,N,C)
    const float* desc2    = (const float*)d_in[3];   // (B,C,H,W)
    const float* homo12   = (const float*)d_in[4];   // (B,3,3)
    float* out = (float*)d_out;

    // workspace layout (floats, then halves; all 16B aligned)
    float* ws   = (float*)d_ws;
    float* acc  = ws;                       // 16 floats: [0]=cnt [1]=fos [2]=sos
    float* a32  = ws + 16;                  // B*N*C
    float* wa32 = a32 + (size_t)BB * NN * CC;
    float* pos  = wa32 + (size_t)BB * NN * CC;
    float* vis  = pos + (size_t)BB * NN;
    _Float16* a16  = (_Float16*)(vis + (size_t)BB * NN);
    _Float16* wa16 = a16 + (size_t)BB * NN * CC;
    _Float16* gd16 = wa16 + (size_t)BB * NN * CC;

    k_init<<<1, 16, 0, stream>>>(acc);

    int pts = BB * NN;
    k_anchor<<<(pts + 255) / 256, 256, 0, stream>>>(kp1_desc, a32, a16);
    k_wa<<<(pts + 255) / 256, 256, 0, stream>>>(kp1, w_kp1, desc2, homo12,
                                                a32, wa32, wa16, pos, vis, acc);
    int cells = BB * MM;
    k_gd<<<(cells + 255) / 256, 256, 0, stream>>>(desc2, gd16);

    dim3 gFos(NN / 16, BB);
    k_fos<<<gFos, 128, 0, stream>>>(a16, gd16, pos, vis, w_kp1, acc);

    dim3 gSos(NN / 16, BB);
    k_sos<<<gSos, 128, 0, stream>>>(wa16, a32, vis, w_kp1, acc);

    k_final<<<1, 32, 0, stream>>>(acc, out);
}
